// DeformConvNet_Encoder256_41188736368691
// MI455X (gfx1250) — matrix operand order repaired and checked by reading, not compiled, not run on hardware
//
#include <hip/hip_runtime.h>
#include <hip/hip_bf16.h>

typedef __attribute__((ext_vector_type(16))) __bf16  v16bf;
typedef __attribute__((ext_vector_type(8)))  float   v8f;
typedef __attribute__((ext_vector_type(4)))  unsigned v4u;   // 16 bytes
typedef __attribute__((ext_vector_type(8)))  unsigned v8u;   // 32 bytes

// float -> bf16 RNE (bit trick)
__device__ __forceinline__ __bf16 f2bf(float f) {
  union { float f; unsigned u; } v; v.f = f;
  unsigned r = v.u + 0x7fffu + ((v.u >> 16) & 1u);
  return __builtin_bit_cast(__bf16, (unsigned short)(r >> 16));
}

// ---------------------------------------------------------------------------
// Pre-pass: f32 NCHW -> bf16 NHWC with channel padding to Cp (power of two).
// ---------------------------------------------------------------------------
__global__ __launch_bounds__(256)
void to_nhwc_bf16(const float* __restrict__ in, __bf16* __restrict__ out,
                  int C, int lgH, int lgW, int lgCp, long total)
{
  const int Cp = 1 << lgCp;
  const int W  = 1 << lgW;
  const int H  = 1 << lgH;
  for (long i = (long)blockIdx.x * blockDim.x + threadIdx.x; i < total;
       i += (long)gridDim.x * blockDim.x) {
    int  ci = (int)(i & (Cp - 1));
    long p  = i >> lgCp;                 // n*H*W + y*W + x
    int  x  = (int)(p & (W - 1));
    long q  = p >> lgW;                  // n*H + y
    int  y  = (int)(q & (H - 1));
    long n  = q >> lgH;
    float v = 0.f;
    if (ci < C) v = in[((n * C + ci) << lgH | y) * (long)W + x];
    out[i] = f2bf(v);
  }
}

// ---------------------------------------------------------------------------
// Pre-pass: weights OIHW f32 -> bf16 [co][Kp], k = s*Cp + ci (s = ky*3+kx),
// zero-padded in both ci (to Cp) and k (to Kp, multiple of 64).
// ---------------------------------------------------------------------------
__global__ __launch_bounds__(256)
void pack_w_bf16(const float* __restrict__ w, __bf16* __restrict__ out,
                 int Ci, int lgCp, int Kp, long total)
{
  const int Cp = 1 << lgCp;
  for (long i = (long)blockIdx.x * blockDim.x + threadIdx.x; i < total;
       i += (long)gridDim.x * blockDim.x) {
    int  t  = (int)(i % Kp);
    long co = i / Kp;
    int  s  = t >> lgCp;
    int  ci = t & (Cp - 1);
    float v = 0.f;
    if (s < 9 && ci < Ci) v = w[(co * Ci + ci) * 9 + s];
    out[i] = f2bf(v);
  }
}

// ---------------------------------------------------------------------------
// Implicit-GEMM 3x3 conv, bf16 WMMA, f32 accumulate.
//   input : bf16 NHWC (channels padded to Cp = 1<<lgCp)
//   weight: bf16 [co][Kp], k = s*Cp + ci, Kp % 64 == 0
//   output: f32 NCHW
// Wave tile: 16 x (16*NCOL); one A fragment feeds NCOL WMMAs.
// A validity = precomputed 9-bit tap mask.
// UNI=true (Cp>=32): a 32-aligned k-block never crosses a tap boundary and
// the tap index is wave-uniform -> one scalar tap offset + one mask test +
// one base address serves both 16B A-groups of the k-step.
// K loop unrolled 2x with explicit ping-pong fragment sets (no reg copies).
// ---------------------------------------------------------------------------
template <int NCOL, bool UNI>
__global__ __launch_bounds__(256)
void conv3x3_wmma(const __bf16* __restrict__ in, const __bf16* __restrict__ wb,
                  float* __restrict__ out,
                  int M, int H, int W, int Co, int Ho, int Wo,
                  int stride, int pad, int lgCp, int Kp, int lgHWo, int lgWo)
{
  const int lane   = threadIdx.x & 31;
  const int wv     = threadIdx.x >> 5;
  const int laneHi = lane >> 4;
  const int laneLo = lane & 15;
  const int Cp     = 1 << lgCp;
  const int hiOff  = 16 * laneHi;

  const int m_base  = (blockIdx.x * 8 + wv) * 16;
  const int co_base = blockIdx.y * (16 * NCOL);

  // ---- A row decode + 9-bit tap-validity mask (hoisted out of K loop) ----
  const int  mr      = m_base + laneLo;
  const bool arow_ok = (mr < M);
  const int  an      = mr >> lgHWo;
  const int  arem    = mr & ((1 << lgHWo) - 1);
  const int  aoy     = arem >> lgWo;
  const int  aox     = arem & ((1 << lgWo) - 1);
  const int  iby     = aoy * stride - pad;
  const int  ibx     = aox * stride - pad;

  unsigned tapmask = 0;
  if (arow_ok) {
#pragma unroll
    for (int s = 0; s < 9; ++s) {
      const int iy = iby + s / 3;
      const int ix = ibx + s % 3;
      if (iy >= 0 && iy < H && ix >= 0 && ix < W) tapmask |= (1u << s);
    }
  }
  // Per-lane base of the top-left tap (only dereferenced when mask allows).
  const __bf16* aptr = in + ((((long)an * H + iby) * W + ibx) << lgCp) + hiOff;

  // ---- B row pointers (halves k0+16*laneHi .. +15 of column laneLo) ----
  const __bf16* wrow[NCOL];
#pragma unroll
  for (int c = 0; c < NCOL; ++c)
    wrow[c] = wb + (long)(co_base + 16 * c + laneLo) * Kp + 16 * laneHi;

  v8f acc[NCOL];
#pragma unroll
  for (int c = 0; c < NCOL; ++c)
    acc[c] = (v8f){0.f, 0.f, 0.f, 0.f, 0.f, 0.f, 0.f, 0.f};

  const v4u z4 = {0u, 0u, 0u, 0u};

  // Load both 16B A-groups of the 32-wide k-step starting at k (32-aligned).
  auto loadApair = [&](int k, v4u& g0, v4u& g1) {
    g0 = z4; g1 = z4;
    if (UNI) {
      // Tap index is uniform for the whole 32-block (Cp >= 32).
      const int s = k >> lgCp;
      if ((tapmask >> s) & 1u) {
        const int ky = (s >= 6) ? 2 : ((s >= 3) ? 1 : 0);
        const int kx = s - 3 * ky;
        const __bf16* p =
            aptr + (((ky * W + kx) << lgCp) + (k & (Cp - 1)));
        g0 = *(const v4u*)(p);
        g1 = *(const v4u*)(p + 8);
      }
    } else {
      // Generic path: the two groups may hit different taps.
#pragma unroll
      for (int g = 0; g < 2; ++g) {
        const int base = k + hiOff + 8 * g;
        const int s    = base >> lgCp;
        const int ci0  = base & (Cp - 1);
        v4u v = z4;
        if ((tapmask >> s) & 1u) {
          const int ky = (s >= 6) ? 2 : ((s >= 3) ? 1 : 0);
          const int kx = s - 3 * ky;
          v = *(const v4u*)(in - hiOff + (aptr - in) +
                            (((ky * W + kx) << lgCp) + ci0));
        }
        if (g == 0) g0 = v; else g1 = v;
      }
    }
  };

  // ---- K loop, 2x unrolled, ping-pong fragment sets (Kp % 64 == 0) ----
  v4u aA0, aA1;
  loadApair(0, aA0, aA1);
  v8u bA[NCOL];
#pragma unroll
  for (int c = 0; c < NCOL; ++c) bA[c] = *(const v8u*)(wrow[c]);

  for (int k0 = 0; k0 < Kp; k0 += 64) {
    // prefetch k0+32 into the B set (always in range: Kp is a multiple of 64)
    v4u aB0, aB1;
    loadApair(k0 + 32, aB0, aB1);
    v8u bB[NCOL];
#pragma unroll
    for (int c = 0; c < NCOL; ++c) bB[c] = *(const v8u*)(wrow[c] + k0 + 32);

    {
      const v16bf a = __builtin_bit_cast(
          v16bf, __builtin_shufflevector(aA0, aA1, 0, 1, 2, 3, 4, 5, 6, 7));
#pragma unroll
      for (int c = 0; c < NCOL; ++c)
        acc[c] = __builtin_amdgcn_wmma_f32_16x16x32_bf16(
            false, a, false, __builtin_bit_cast(v16bf, bA[c]),
            (short)0, acc[c], false, false);
    }

    // prefetch k0+64 into the A set
    const int kn = k0 + 64;
    if (kn < Kp) {
      loadApair(kn, aA0, aA1);
#pragma unroll
      for (int c = 0; c < NCOL; ++c) bA[c] = *(const v8u*)(wrow[c] + kn);
    }

    {
      const v16bf a = __builtin_bit_cast(
          v16bf, __builtin_shufflevector(aB0, aB1, 0, 1, 2, 3, 4, 5, 6, 7));
#pragma unroll
      for (int c = 0; c < NCOL; ++c)
        acc[c] = __builtin_amdgcn_wmma_f32_16x16x32_bf16(
            false, a, false, __builtin_bit_cast(v16bf, bB[c]),
            (short)0, acc[c], false, false);
    }
  }

  // ---- store D (f32 NCHW); decode m once per row r ----
#pragma unroll
  for (int r = 0; r < 8; ++r) {
    const int m = m_base + r + (laneHi ? 8 : 0);
    if (m < M) {
      const int  n     = m >> lgHWo;
      const int  rem   = m & ((1 << lgHWo) - 1);
      const int  oy    = rem >> lgWo, ox = rem & ((1 << lgWo) - 1);
      const long pbase = (((long)n * Co) * Ho + oy) * Wo + ox;
      const long cstep = (long)Ho * Wo;
#pragma unroll
      for (int c = 0; c < NCOL; ++c)
        out[pbase + (co_base + 16 * c + laneLo) * cstep] = acc[c][r];
    }
  }
}

// ---------------------------------------------------------------------------
// BatchNorm pass 1: per-channel sum / sumsq over (N,H,W).
// ---------------------------------------------------------------------------
__global__ __launch_bounds__(256)
void bn_stats(const float* __restrict__ x, float* __restrict__ stats,
              int N, int C, int HW)
{
  __shared__ float ss[256];
  __shared__ float sq[256];
  const int c = blockIdx.x;
  const long total = (long)N * HW;
  float s = 0.f, q = 0.f;
  for (long i = (long)blockIdx.y * blockDim.x + threadIdx.x; i < total;
       i += (long)gridDim.y * blockDim.x) {
    long n = i / HW, p = i % HW;
    float v = x[(n * C + c) * (long)HW + p];
    s += v; q += v * v;
  }
  ss[threadIdx.x] = s; sq[threadIdx.x] = q;
  __syncthreads();
  for (int o = 128; o > 0; o >>= 1) {
    if ((int)threadIdx.x < o) {
      ss[threadIdx.x] += ss[threadIdx.x + o];
      sq[threadIdx.x] += sq[threadIdx.x + o];
    }
    __syncthreads();
  }
  if (threadIdx.x == 0) {
    atomicAdd(&stats[c], ss[0]);
    atomicAdd(&stats[C + c], sq[0]);
  }
}

// ---------------------------------------------------------------------------
// BatchNorm pass 2 + activation (slope=0 -> ReLU, 0.2 -> LeakyReLU), in-place.
// ---------------------------------------------------------------------------
__global__ __launch_bounds__(256)
void bn_apply(float* __restrict__ x, const float* __restrict__ stats,
              const float* __restrict__ g, const float* __restrict__ b,
              int C, int HW, long total, float inv_cnt, float slope)
{
  for (long i = (long)blockIdx.x * blockDim.x + threadIdx.x; i < total;
       i += (long)gridDim.x * blockDim.x) {
    int   c   = (int)((i / HW) % C);
    float m   = stats[c] * inv_cnt;
    float var = stats[C + c] * inv_cnt - m * m;
    float y   = (x[i] - m) * rsqrtf(var + 1e-5f) * g[c] + b[c];
    x[i] = (y > 0.f) ? y : y * slope;
  }
}

// ---------------------------------------------------------------------------
// ConvOffset2D bilinear resample (faithful to the (b*c,h,w,2) reinterpret).
// ---------------------------------------------------------------------------
__global__ __launch_bounds__(256)
void deform_sample(const float* __restrict__ x, const float* __restrict__ off,
                   float* __restrict__ out, int H, int W, long total)
{
  const long HW = (long)H * W;
  for (long i = (long)blockIdx.x * blockDim.x + threadIdx.x; i < total;
       i += (long)gridDim.x * blockDim.x) {
    long bc = i / HW;
    long p  = i % HW;
    int  yy = (int)(p / W), xx = (int)(p % W);
    float cy = off[2 * i]     + (float)yy;
    float cx = off[2 * i + 1] + (float)xx;
    cy = fminf(fmaxf(cy, 0.f), (float)(H - 1));
    cx = fminf(fmaxf(cx, 0.f), (float)(W - 1));
    float y0 = floorf(cy), y1 = ceilf(cy);
    float x0 = floorf(cx), x1 = ceilf(cx);
    float dy = cy - y0, dx = cx - x0;
    const float* xp = x + bc * HW;
    float v00 = xp[(long)(int)y0 * W + (int)x0];
    float v01 = xp[(long)(int)y0 * W + (int)x1];
    float v10 = xp[(long)(int)y1 * W + (int)x0];
    float v11 = xp[(long)(int)y1 * W + (int)x1];
    out[i] = v00 * (1.f - dy) * (1.f - dx) + v01 * (1.f - dy) * dx +
             v10 * dy * (1.f - dx)         + v11 * dy * dx;
  }
}

// ---------------------------------------------------------------------------
// Final 1x1 conv (32x100x1024) + sigmoid.
// ---------------------------------------------------------------------------
__global__ __launch_bounds__(256)
void head_sigmoid(const float* __restrict__ h, const float* __restrict__ w,
                  const float* __restrict__ bias, float* __restrict__ out,
                  int B, int NZ, int K)
{
  int i = blockIdx.x * blockDim.x + threadIdx.x;
  if (i >= B * NZ) return;
  int b = i / NZ, z = i % NZ;
  float s = bias[z];
  for (int k = 0; k < K; ++k) s += h[(long)b * K + k] * w[(long)z * K + k];
  out[i] = 1.f / (1.f + expf(-s));
}

// ---------------------------------------------------------------------------
// Host-side orchestration
// ---------------------------------------------------------------------------
static inline int ilog2i(int v) { int l = 0; while ((1 << l) < v) ++l; return l; }

struct ConvBufs { __bf16* act; __bf16* wgt; };

static inline void launch_conv(const float* in, const float* w, float* out,
                               ConvBufs bufs,
                               int N, int Ci, int H, int W, int Co,
                               int Ho, int Wo, int s, int p, hipStream_t st)
{
  const int Cp    = (Ci < 8) ? 8 : Ci;           // all Ci are pow2 or 3->8
  const int lgCp  = ilog2i(Cp);
  const int Kp    = (9 * Cp + 63) & ~63;         // multiple of 64 for 2x unroll
  const int lgH   = ilog2i(H);
  const int lgW   = ilog2i(W);
  const int lgWo  = ilog2i(Wo);
  const int lgHWo = ilog2i(Ho * Wo);
  const bool uni  = (lgCp >= 5);                 // 32-block never crosses a tap

  // input -> bf16 NHWC (padded)
  {
    long total = ((long)N * H * W) << lgCp;
    int blocks = (int)((total + 255) / 256);
    if (blocks > 8192) blocks = 8192;
    to_nhwc_bf16<<<blocks, 256, 0, st>>>(in, bufs.act, Ci, lgH, lgW, lgCp, total);
  }
  // weights -> bf16 [co][Kp]
  {
    long total = (long)Co * Kp;
    int blocks = (int)((total + 255) / 256);
    if (blocks > 4096) blocks = 4096;
    pack_w_bf16<<<blocks, 256, 0, st>>>(w, bufs.wgt, Ci, lgCp, Kp, total);
  }
  // WMMA conv: widest column tile Co supports; UNI path when Cp >= 32.
  {
    const int M = N * Ho * Wo;
    const int gx = (M + 127) / 128;
#define CONV_DISPATCH(NC)                                                      \
    do {                                                                       \
      dim3 grid(gx, Co / (16 * (NC)));                                         \
      if (uni)                                                                 \
        conv3x3_wmma<NC, true><<<grid, 256, 0, st>>>(                          \
            bufs.act, bufs.wgt, out, M, H, W, Co, Ho, Wo, s, p,                \
            lgCp, Kp, lgHWo, lgWo);                                            \
      else                                                                     \
        conv3x3_wmma<NC, false><<<grid, 256, 0, st>>>(                         \
            bufs.act, bufs.wgt, out, M, H, W, Co, Ho, Wo, s, p,                \
            lgCp, Kp, lgHWo, lgWo);                                            \
    } while (0)
    if      (Co % 64 == 0) CONV_DISPATCH(4);
    else if (Co % 32 == 0) CONV_DISPATCH(2);
    else                   CONV_DISPATCH(1);
#undef CONV_DISPATCH
  }
}

static inline void run_bn(float* xbuf, float* stats, const float* g,
                          const float* b, int N, int C, int HW, float slope,
                          hipStream_t st)
{
  hipMemsetAsync(stats, 0, sizeof(float) * 2 * (size_t)C, st);
  long cnt = (long)N * HW;
  int chunks = (int)((cnt + 255) / 256);
  if (chunks > 128) chunks = 128;
  if (chunks < 1)   chunks = 1;
  dim3 g2((unsigned)C, (unsigned)chunks);
  bn_stats<<<g2, 256, 0, st>>>(xbuf, stats, N, C, HW);

  long total = (long)N * C * HW;
  int blocks = (int)((total + 255) / 256);
  if (blocks > 8192) blocks = 8192;
  bn_apply<<<blocks, 256, 0, st>>>(xbuf, stats, g, b, C, HW, total,
                                   1.0f / (float)cnt, slope);
}

extern "C" void kernel_launch(void* const* d_in, const int* in_sizes, int n_in,
                              void* d_out, int out_size, void* d_ws, size_t ws_size,
                              hipStream_t stream)
{
  (void)in_sizes; (void)n_in; (void)out_size; (void)ws_size;

  const float* x       = (const float*)d_in[0];
  const float* w_first = (const float*)d_in[1];
  const float* g_first = (const float*)d_in[2];
  const float* b_first = (const float*)d_in[3];
  const float* w_last  = (const float*)d_in[24];
  const float* g_last  = (const float*)d_in[25];
  const float* b_last  = (const float*)d_in[26];
  const float* w_out   = (const float*)d_in[27];
  const float* b_out   = (const float*)d_in[28];

  // Workspace layout (bytes):
  //   [0,          33554432)   bufX   : f32 activations
  //   [33554432,   67108864)   bufY   : f32 deformed activations
  //   [67108864,  134217728)   bufOff : f32 offset-conv output (layer0 peak)
  //   [134217728, 167772160)   actbf  : bf16 NHWC conv input (peak: padded x)
  //   [167772160, 177209344)   wbf    : bf16 packed weights (peak: w_last)
  //   [177209344, 177217536)   stats  : 2*1024 f32
  char* ws = (char*)d_ws;
  float*  bufX   = (float*)(ws);
  float*  bufY   = (float*)(ws + 33554432u);
  float*  bufOff = (float*)(ws + 67108864u);
  __bf16* actbf  = (__bf16*)(ws + 134217728u);
  __bf16* wbf    = (__bf16*)(ws + 167772160u);
  float*  stats  = (float*)(ws + 177209344u);
  ConvBufs cb{actbf, wbf};

  const int N = 32;
  const int FEATS[7] = {16, 32, 64, 128, 256, 512, 1024};

  // ---- first conv: 3 -> 16, 256 -> 128, BN + ReLU ----
  launch_conv(x, w_first, bufX, cb, N, 3, 256, 256, 16, 128, 128, 2, 1, stream);
  run_bn(bufX, stats, g_first, b_first, N, 16, 128 * 128, 0.0f, stream);

  // ---- 5 deform layers ----
  int Hi = 128;
  for (int i = 0; i < 5; ++i) {
    const int ci = FEATS[i], co = FEATS[i + 1];
    const float* ow = (const float*)d_in[4 + 4 * i];
    const float* cw = (const float*)d_in[5 + 4 * i];
    const float* gg = (const float*)d_in[6 + 4 * i];
    const float* bb = (const float*)d_in[7 + 4 * i];
    const int Ho = Hi / 2;

    // offset conv: ci -> 2ci, stride 1, pad 1 (raw conv output)
    launch_conv(bufX, ow, bufOff, cb, N, ci, Hi, Hi, 2 * ci, Hi, Hi, 1, 1, stream);

    // bilinear deform sampling
    {
      long total = (long)N * ci * Hi * Hi;
      int blocks = (int)((total + 255) / 256);
      if (blocks > 8192) blocks = 8192;
      deform_sample<<<blocks, 256, 0, stream>>>(bufX, bufOff, bufY, Hi, Hi, total);
    }

    // main conv: ci -> co, stride 2, pad 1, then BN + LeakyReLU(0.2)
    launch_conv(bufY, cw, bufX, cb, N, ci, Hi, Hi, co, Ho, Ho, 2, 1, stream);
    run_bn(bufX, stats, gg, bb, N, co, Ho * Ho, 0.2f, stream);

    Hi = Ho;
  }

  // ---- last conv: 512 -> 1024, 4x4 -> 1x1 (stride 2, pad 0), BN + ReLU ----
  launch_conv(bufX, w_last, bufY, cb, N, 512, 4, 4, 1024, 1, 1, 2, 0, stream);
  run_bn(bufY, stats, g_last, b_last, N, 1024, 1, 0.0f, stream);

  // ---- 1x1 conv to NZ=100 + sigmoid ----
  head_sigmoid<<<(32 * 100 + 255) / 256, 256, 0, stream>>>(
      bufY, w_out, b_out, (float*)d_out, 32, 100, 1024);
}
